// LTSGNN_76115410420106
// MI455X (gfx1250) — matrix operand order repaired and checked by reading, not compiled, not run on hardware
//
#include <hip/hip_runtime.h>
#include <hip/hip_bf16.h>
#include <cstdint>
#include <cstddef>

#define NN 100000
#define EE 3200000
#define GG 64

typedef __attribute__((ext_vector_type(16))) _Float16 v16h;
typedef __attribute__((ext_vector_type(8)))  _Float16 v8h;
typedef __attribute__((ext_vector_type(8)))  float    v8f;

// ---------- helpers ----------
static __device__ __forceinline__ unsigned f32_ord(float x) {
  unsigned u = __float_as_uint(x);
  return (u & 0x80000000u) ? ~u : (u | 0x80000000u);
}
static __device__ __forceinline__ float ord_f32(unsigned e) {
  return (e & 0x80000000u) ? __uint_as_float(e & 0x7FFFFFFFu) : __uint_as_float(~e);
}
// edge j endpoints; j < E: real edge, else self-loop (i,i) with i=(j-E)%N
static __device__ __forceinline__ void edge_ep(int j, const int* __restrict__ srcE,
                                               const int* __restrict__ dstE,
                                               int E, int N, int& s, int& d) {
  if (j < E) { s = srcE[j]; d = dstE[j]; }
  else       { int i = (j - E) % N; s = i; d = i; }
}

// ---------- fill ----------
__global__ void k_fill_f32(float* p, float v, int n) {
  int t = blockIdx.x * blockDim.x + threadIdx.x;
  if (t < n) p[t] = v;
}
__global__ void k_fill_u32(unsigned* p, unsigned v, int n) {
  int t = blockIdx.x * blockDim.x + threadIdx.x;
  if (t < n) p[t] = v;
}

// ---------- degree + self-loop attr accumulation over INPUT edge list ----------
__global__ void k_deg_loop(const int* __restrict__ srcE, const int* __restrict__ dstE,
                           const float* __restrict__ ea_in, float* __restrict__ deg,
                           float* __restrict__ loopsum, int E, int e_in, int N, int ed) {
  int j = blockIdx.x * blockDim.x + threadIdx.x;
  if (j >= e_in) return;
  int s, d; edge_ep(j, srcE, dstE, E, N, s, d);
  atomicAdd(&deg[d], 1.0f);
  for (int k = 0; k < ed; ++k)
    atomicAdd(&loopsum[d * ed + k], ea_in[(size_t)j * ed + k]);
}

// ---------- wea[k] = sum_c We[k,c]*a_e[c]  (tiny) ----------
__global__ void k_wea(const float* __restrict__ We, const float* __restrict__ aev,
                      float* __restrict__ wea, int ed, int C) {
  int k = threadIdx.x;
  if (k >= ed) return;
  float s = 0.f;
  for (int c = 0; c < C; ++c) s += We[k * C + c] * aev[c];
  wea[k] = s;
}

// ---------- layer-1 dense: h = x(N,5) @ W(5,64) ----------
__global__ void k_dense5(const float* __restrict__ x, const float* __restrict__ W,
                         float* __restrict__ h, int N) {
  int t = blockIdx.x * blockDim.x + threadIdx.x;
  if (t >= N * 64) return;
  int n = t >> 6, c = t & 63;
  const float* xr = x + n * 5;
  float s = 0.f;
#pragma unroll
  for (int k = 0; k < 5; ++k) s += xr[k] * W[k * 64 + c];
  h[t] = s;
}

// ---------- pack W(64,C) f32 into WMMA B-fragment layout, f16 ----------
// Bpack[((ct*2+q)*32 + lane)*16 + i] = W[(q*32 + (lane>=16?8:0) + (i>=8?8:0) + i)*C + ct*16+(lane&15)]
// => each lane's 16 halves for one (ctile,kchunk) are 32 contiguous bytes.
__global__ void k_wpack(const float* __restrict__ W, _Float16* __restrict__ Bpack, int C) {
  int t = blockIdx.x * blockDim.x + threadIdx.x;
  if (t >= C * 64) return;
  int i  = t & 15;
  int L  = (t >> 4) & 31;
  int q  = (t >> 9) & 1;
  int ct = t >> 10;
  int bcol = ct * 16 + (L & 15);
  int kb0  = q * 32 + ((L >= 16) ? 8 : 0) + ((i >= 8) ? 8 : 0);
  Bpack[t] = (_Float16)W[(kb0 + i) * C + bcol];
}

// ---------- WMMA GEMM: H(NPAD,C) = A16(NPAD,64) @ W(64,C) ----------
// block = (C/16) waves; each wave owns one 16-col tile, keeps both B fragments
// in registers, and sweeps 4 row tiles (64 rows per block). No divergence:
// EXEC is all ones at every WMMA; stores go to the padded H buffer.
template <int C>
__global__ void k_gemm_wmma(const _Float16* __restrict__ A16,
                            const _Float16* __restrict__ Bpack,
                            float* __restrict__ H) {
  constexpr int MT = 4;
  const int ct    = threadIdx.x >> 5;
  const int lane  = threadIdx.x & 31;
  const int rlo   = lane & 15;
  const bool hi   = lane >= 16;
  const int kbase = hi ? 8 : 0;

  const v16h b0 = *(const v16h*)(Bpack + ((size_t)(ct * 2 + 0) * 32 + lane) * 16);
  const v16h b1 = *(const v16h*)(Bpack + ((size_t)(ct * 2 + 1) * 32 + lane) * 16);

  const int col  = ct * 16 + rlo;
  const int row0 = blockIdx.x * (16 * MT);

#pragma unroll
  for (int mt = 0; mt < MT; ++mt) {
    const int m0 = row0 + mt * 16;
    const _Float16* ap = A16 + (size_t)(m0 + rlo) * 64 + kbase;
    v8h a0lo = *(const v8h*)(ap);          // K = kbase+0..7
    v8h a0hi = *(const v8h*)(ap + 16);     // K = kbase+16..23
    v8h a1lo = *(const v8h*)(ap + 32);     // kk=32 chunk
    v8h a1hi = *(const v8h*)(ap + 48);
    v16h a0 = __builtin_shufflevector(a0lo, a0hi, 0,1,2,3,4,5,6,7,8,9,10,11,12,13,14,15);
    v16h a1 = __builtin_shufflevector(a1lo, a1hi, 0,1,2,3,4,5,6,7,8,9,10,11,12,13,14,15);
    v8f acc = {};
    acc = __builtin_amdgcn_wmma_f32_16x16x32_f16(false, a0, false, b0,
                                                 (short)0, acc, false, false);
    acc = __builtin_amdgcn_wmma_f32_16x16x32_f16(false, a1, false, b1,
                                                 (short)0, acc, false, false);
    // D layout: VGPR v -> row m0+v (lanes 0-15) / m0+v+8 (lanes 16-31)
    float* hp = H + (size_t)(m0 + (hi ? 8 : 0)) * C + col;
#pragma unroll
    for (int v = 0; v < 8; ++v) hp[v * C] = acc[v];
  }
}

// ---------- asrc[n] = h[n,:]·a_src, adst likewise ----------
__global__ void k_attn_vec(const float* __restrict__ h, const float* __restrict__ avs,
                           const float* __restrict__ avd, float* __restrict__ asrc,
                           float* __restrict__ adst, int N, int C) {
  int n = blockIdx.x * blockDim.x + threadIdx.x;
  if (n >= N) return;
  const float* hr = h + (size_t)n * C;
  float s = 0.f, d = 0.f;
  for (int c = 0; c < C; ++c) { float v = hr[c]; s += v * avs[c]; d += v * avd[c]; }
  asrc[n] = s; adst[n] = d;
}

// ---------- raw attention + segment max (monotone-uint atomicMax) ----------
__global__ void k_alpha_max(const int* __restrict__ srcE, const int* __restrict__ dstE,
                            const float* __restrict__ ea_in, const float* __restrict__ loopsum,
                            const float* __restrict__ deg, const float* __restrict__ wea,
                            const float* __restrict__ asrc, const float* __restrict__ adst,
                            float* __restrict__ alpha, unsigned* __restrict__ amaxU,
                            int E, int e_in, int e_total, int N, int ed) {
  int j = blockIdx.x * blockDim.x + threadIdx.x;
  if (j >= e_total) return;
  int s, d; edge_ep(j, srcE, dstE, E, N, s, d);
  float ae = 0.f;
  if (j < e_in) {
    for (int k = 0; k < ed; ++k) ae += ea_in[(size_t)j * ed + k] * wea[k];
  } else {                                   // fresh self-loop: mean of incoming attrs
    int i = j - e_in;
    float invd = 1.0f / fmaxf(deg[i], 1.0f);
    for (int k = 0; k < ed; ++k) ae += loopsum[i * ed + k] * invd * wea[k];
  }
  float r = asrc[s] + adst[d] + ae;
  r = (r > 0.f) ? r : 0.2f * r;              // leaky_relu 0.2
  alpha[j] = r;
  atomicMax(&amaxU[d], f32_ord(r));
}

// ---------- exp(raw - max) + segment sum ----------
__global__ void k_ex_denom(const int* __restrict__ srcE, const int* __restrict__ dstE,
                           float* __restrict__ alpha, const unsigned* __restrict__ amaxU,
                           float* __restrict__ denom, int E, int e_total, int N) {
  int j = blockIdx.x * blockDim.x + threadIdx.x;
  if (j >= e_total) return;
  int s, d; edge_ep(j, srcE, dstE, E, N, s, d);
  float ex = __expf(alpha[j] - ord_f32(amaxU[d]));
  alpha[j] = ex;
  atomicAdd(&denom[d], ex);
}

// ---------- normalize (alpha becomes next layer's edge attr) ----------
__global__ void k_norm(const int* __restrict__ srcE, const int* __restrict__ dstE,
                       float* __restrict__ alpha, const float* __restrict__ denom,
                       int E, int e_total, int N) {
  int j = blockIdx.x * blockDim.x + threadIdx.x;
  if (j >= e_total) return;
  int s, d; edge_ep(j, srcE, dstE, E, N, s, d);
  alpha[j] = alpha[j] / (denom[d] + 1e-16f);
}

// ---------- out[dst,:] += alpha * h[src,:]  (one thread per edge x 4 channels) ----------
__global__ void k_scatter(const int* __restrict__ srcE, const int* __restrict__ dstE,
                          const float* __restrict__ alpha, const float* __restrict__ h,
                          float* __restrict__ outacc, int E, int e_total, int N, int C) {
  long t = blockIdx.x * (long)blockDim.x + threadIdx.x;
  int chunks = C >> 2;
  long tot = (long)e_total * chunks;
  if (t >= tot) return;
  int j  = (int)(t / chunks);
  int c0 = ((int)(t - (long)j * chunks)) << 2;
  int s, d; edge_ep(j, srcE, dstE, E, N, s, d);
  float a = alpha[j];
  const float4 hv = *(const float4*)(h + (size_t)s * C + c0);
  float* o = outacc + (size_t)d * C + c0;
  atomicAdd(o + 0, a * hv.x);
  atomicAdd(o + 1, a * hv.y);
  atomicAdd(o + 2, a * hv.z);
  atomicAdd(o + 3, a * hv.w);
}

// ---------- bias + relu -> f16 features for next layer's WMMA (zero pad rows) ----------
__global__ void k_bias_relu_h16(const float* __restrict__ outacc, const float* __restrict__ b,
                                _Float16* __restrict__ xh, int N, int NPAD, int C) {
  int t = blockIdx.x * blockDim.x + threadIdx.x;
  if (t >= NPAD * C) return;
  int n = t / C, c = t - n * C;
  float v = 0.f;
  if (n < N) v = fmaxf(outacc[(size_t)n * C + c] + b[c], 0.f);
  xh[t] = (_Float16)v;
}

// ---------- final bias + global mean pool ----------
__global__ void k_bias_pool(const float* __restrict__ outacc, const float* __restrict__ b,
                            const int* __restrict__ batch, float* __restrict__ pooled,
                            float* __restrict__ cnt, int N, int C) {
  int t = blockIdx.x * blockDim.x + threadIdx.x;
  if (t >= N * C) return;
  int n = t / C, c = t - n * C;
  int g = batch[n];
  atomicAdd(&pooled[g * C + c], outacc[t] + b[c]);
  if (c == 0) atomicAdd(&cnt[g], 1.0f);
}
__global__ void k_pool_final(const float* __restrict__ pooled, const float* __restrict__ cnt,
                             float* __restrict__ out, int G, int C) {
  int t = blockIdx.x * blockDim.x + threadIdx.x;
  if (t >= G * C) return;
  int g = t / C;
  out[t] = pooled[t] / fmaxf(cnt[g], 1.0f);
}

// =================== host orchestration ===================
static inline unsigned gdiv(long n, int b) { return (unsigned)((n + b - 1) / b); }

extern "C" void kernel_launch(void* const* d_in, const int* in_sizes, int n_in,
                              void* d_out, int out_size, void* d_ws, size_t ws_size,
                              hipStream_t stream) {
  const int N = NN, E = EE, G = GG;
  const int NPAD = ((N + 63) / 64) * 64;   // rows padded to 4 WMMA tiles
  const int ETMAX = E + 3 * N;

  const float* x     = (const float*)d_in[0];
  const int*   ei    = (const int*)d_in[1];
  const float* eattr = (const float*)d_in[2];
  const int*   batch = (const int*)d_in[3];
  const float* W[3]  = { (const float*)d_in[4],  (const float*)d_in[10], (const float*)d_in[16] };
  const float* AS[3] = { (const float*)d_in[5],  (const float*)d_in[11], (const float*)d_in[17] };
  const float* AD[3] = { (const float*)d_in[6],  (const float*)d_in[12], (const float*)d_in[18] };
  const float* WE[3] = { (const float*)d_in[7],  (const float*)d_in[13], (const float*)d_in[19] };
  const float* AE[3] = { (const float*)d_in[8],  (const float*)d_in[14], (const float*)d_in[20] };
  const float* B[3]  = { (const float*)d_in[9],  (const float*)d_in[15], (const float*)d_in[21] };
  const int* srcE = ei;
  const int* dstE = ei + E;

  // ---- carve workspace ----
  char* ws = (char*)d_ws;
  size_t off = 0;
  auto carve = [&](size_t bytes) -> void* {
    void* p = ws + off;
    off = (off + bytes + 255) & ~(size_t)255;
    return p;
  };
  float*     hbuf    = (float*)carve((size_t)NPAD * 64 * 4);
  float*     outacc  = (float*)carve((size_t)N * 64 * 4);
  _Float16*  xh      = (_Float16*)carve((size_t)NPAD * 64 * 2);
  _Float16*  bpack   = (_Float16*)carve((size_t)64 * 64 * 2);
  float*     asrc    = (float*)carve((size_t)N * 4);
  float*     adst    = (float*)carve((size_t)N * 4);
  float*     deg     = (float*)carve((size_t)N * 4);
  float*     loopsum = (float*)carve((size_t)N * 3 * 4);
  unsigned*  amaxU   = (unsigned*)carve((size_t)N * 4);
  float*     denom   = (float*)carve((size_t)N * 4);
  float*     alphaA  = (float*)carve((size_t)ETMAX * 4);
  float*     alphaB  = (float*)carve((size_t)ETMAX * 4);
  float*     wea     = (float*)carve(64);
  float*     pooled  = (float*)carve((size_t)G * 32 * 4);
  float*     cnt     = (float*)carve((size_t)G * 4);
  (void)ws_size; (void)n_in; (void)in_sizes; (void)out_size;

  const int BS = 256;
  const unsigned ORD_NEG_INF = 0x007FFFFFu; // f32_ord(-inf)

  const int Cin[3] = { 5, 64, 64 };
  const int C[3]   = { 64, 64, 32 };
  const int ED[3]  = { 3, 1, 1 };
  const float* EAin[3] = { eattr, alphaA, alphaB };
  float*       AOut[3] = { alphaA, alphaB, alphaA };

  for (int l = 0; l < 3; ++l) {
    const int e_in = E + l * N;
    const int e_total = e_in + N;
    const int c = C[l], ed = ED[l];

    // zero accumulators / init segment-max
    k_fill_f32<<<gdiv(N, BS), BS, 0, stream>>>(deg, 0.f, N);
    k_fill_f32<<<gdiv((long)N * ed, BS), BS, 0, stream>>>(loopsum, 0.f, N * ed);
    k_fill_f32<<<gdiv(N, BS), BS, 0, stream>>>(denom, 0.f, N);
    k_fill_f32<<<gdiv((long)N * c, BS), BS, 0, stream>>>(outacc, 0.f, N * c);
    k_fill_u32<<<gdiv(N, BS), BS, 0, stream>>>(amaxU, ORD_NEG_INF, N);

    // self-loop attr = mean of incoming edge attrs
    k_deg_loop<<<gdiv(e_in, BS), BS, 0, stream>>>(srcE, dstE, EAin[l], deg, loopsum,
                                                  E, e_in, N, ed);
    // wea = We @ a_e
    k_wea<<<1, 32, 0, stream>>>(WE[l], AE[l], wea, ed, c);

    // h = x @ W
    if (l == 0) {
      k_dense5<<<gdiv((long)N * 64, BS), BS, 0, stream>>>(x, W[0], hbuf, N);
    } else {
      k_wpack<<<gdiv((long)c * 64, BS), BS, 0, stream>>>(W[l], bpack, c);
      if (c == 64)
        k_gemm_wmma<64><<<dim3(NPAD / 64), dim3(128), 0, stream>>>(xh, bpack, hbuf);
      else
        k_gemm_wmma<32><<<dim3(NPAD / 64), dim3(64), 0, stream>>>(xh, bpack, hbuf);
    }

    // per-node attention projections
    k_attn_vec<<<gdiv(N, BS), BS, 0, stream>>>(hbuf, AS[l], AD[l], asrc, adst, N, c);

    // segment softmax over destination
    k_alpha_max<<<gdiv(e_total, BS), BS, 0, stream>>>(srcE, dstE, EAin[l], loopsum, deg,
                                                      wea, asrc, adst, AOut[l], amaxU,
                                                      E, e_in, e_total, N, ed);
    k_ex_denom<<<gdiv(e_total, BS), BS, 0, stream>>>(srcE, dstE, AOut[l], amaxU, denom,
                                                     E, e_total, N);
    k_norm<<<gdiv(e_total, BS), BS, 0, stream>>>(srcE, dstE, AOut[l], denom,
                                                 E, e_total, N);
    // aggregate
    k_scatter<<<gdiv((long)e_total * (c >> 2), BS), BS, 0, stream>>>(
        srcE, dstE, AOut[l], hbuf, outacc, E, e_total, N, c);

    if (l < 2) {
      // bias + relu, convert to f16 for the next WMMA stage (pad rows zeroed)
      k_bias_relu_h16<<<gdiv((long)NPAD * c, BS), BS, 0, stream>>>(outacc, B[l], xh,
                                                                   N, NPAD, c);
    } else {
      // final bias + global mean pool
      k_fill_f32<<<gdiv((long)G * c, BS), BS, 0, stream>>>(pooled, 0.f, G * c);
      k_fill_f32<<<1, G, 0, stream>>>(cnt, 0.f, G);
      k_bias_pool<<<gdiv((long)N * c, BS), BS, 0, stream>>>(outacc, B[l], batch,
                                                            pooled, cnt, N, c);
      k_pool_final<<<gdiv((long)G * c, BS), BS, 0, stream>>>(pooled, cnt,
                                                             (float*)d_out, G, c);
    }
  }
}
